// FlowPre_50105088475269
// MI455X (gfx1250) — compile-verified, matrix operand order checked by reading, not run on hardware
//
#include <hip/hip_runtime.h>
#include <hip/hip_bf16.h>
#include <vector>

// ---------------- problem constants ----------------
constexpr int BATCH = 16384;
constexpr int DDIM  = 128;
constexpr int HALF  = 64;
constexpr int CTXW  = 8;
constexpr int HID   = 256;
constexpr int AIN   = HALF + CTXW;   // 72
constexpr int KP1   = 96;            // AIN padded to multiple of 32
constexpr int AOUT  = 2 * HALF;      // 128
constexpr int ROUT  = HALF * 23;     // 1472
constexpr int NLU   = 22;
constexpr int NAF   = 6;
constexpr int NRQ   = 15;
constexpr int CH    = 2048;          // row chunk for coupling layers
constexpr float TBND = 3.0f;
constexpr float MBW  = 0.001f;
constexpr float MDC  = 0.001f;

typedef __attribute__((ext_vector_type(16))) _Float16 v16h;
typedef __attribute__((ext_vector_type(8)))  _Float16 v8h;
typedef __attribute__((ext_vector_type(8)))  float    v8f;
typedef __attribute__((ext_vector_type(4)))  float    v4f;

__device__ __forceinline__ float softplusf(float x) {
    return (x > 20.f) ? x : log1pf(expf(x));
}

// ---------------- WMMA fragment loaders ----------------
// 16-bit A-matrix 16x32 layout (ISA 7.12.2): lane holds row M=lane&15;
// lanes 0-15 hold K = k0+[0..7] and k0+[16..23]; lanes 16-31 hold
// K = k0+[8..15] and k0+[24..31].

// pure f16 source: two 16B loads per fragment
__device__ __forceinline__ v16h load_frag_f16(const _Float16* __restrict__ base,
                                              int ldk, int row0, int k0, int lane) {
    int r  = row0 + (lane & 15);
    int kb = k0 + 8 * (lane >> 4);
    const _Float16* p = base + (long)r * ldk + kb;
    v8h lo = *(const v8h*)(p);
    v8h hi = *(const v8h*)(p + 16);
    v16h f;
#pragma unroll
    for (int i = 0; i < 8; i++) { f[i] = lo[i]; f[8 + i] = hi[i]; }
    return f;
}

// f32 source with on-the-fly convert (used for LU GEMM reading f32 state)
__device__ __forceinline__ v16h load_frag_f32(const float* __restrict__ base,
                                              int ldk, int row0, int k0, int lane) {
    int r  = row0 + (lane & 15);
    int kb = k0 + 8 * (lane >> 4);
    const float* p = base + (long)r * ldk + kb;
    v4f a0 = *(const v4f*)(p);
    v4f a1 = *(const v4f*)(p + 4);
    v4f b0 = *(const v4f*)(p + 16);
    v4f b1 = *(const v4f*)(p + 20);
    v16h f;
#pragma unroll
    for (int i = 0; i < 4; i++) {
        f[i]      = (_Float16)a0[i];
        f[4 + i]  = (_Float16)a1[i];
        f[8 + i]  = (_Float16)b0[i];
        f[12 + i] = (_Float16)b1[i];
    }
    return f;
}

// ---------------- generic WMMA GEMM ----------------
// C[rows x N] = act(A[rows x K] @ W[N x K]^T + bias)
// One wave computes an MT x NT grid of 16x16 tiles: B fragments are loaded
// once per K-step and reused across MT A fragments; each A fragment feeds
// NT WMMAs.  MT=2, NT=4 -> 8 v_wmma per K iteration, 32x64 output per wave.
template <int RELU, int AF32, int OUTF16, int MT, int NT>
__global__ __launch_bounds__(128)
void gemm_wmma(const void* __restrict__ Av, const _Float16* __restrict__ W,
               const float* __restrict__ bias, void* __restrict__ Cv,
               int K, int N) {
    int lane = threadIdx.x & 31;
    int wave = threadIdx.x >> 5;
    int row0 = (blockIdx.x * 4 + wave) * (16 * MT);
    int col0 = blockIdx.y * (16 * NT);
    v8f acc[MT][NT];
#pragma unroll
    for (int m = 0; m < MT; m++)
#pragma unroll
        for (int n = 0; n < NT; n++) acc[m][n] = (v8f){};

    for (int k0 = 0; k0 < K; k0 += 32) {
        v16h bf[NT];
#pragma unroll
        for (int n = 0; n < NT; n++)
            bf[n] = load_frag_f16(W, K, col0 + 16 * n, k0, lane);
        if (k0 + 32 < K) {  // prefetch next K-tile of the weight operand
            const _Float16* wn = W + (long)(col0 + (lane & 15)) * K + k0 + 32;
            __builtin_prefetch((const void*)wn, 0, 1);
        }
#pragma unroll
        for (int m = 0; m < MT; m++) {
            v16h a = AF32
                ? load_frag_f32((const float*)Av, K, row0 + 16 * m, k0, lane)
                : load_frag_f16((const _Float16*)Av, K, row0 + 16 * m, k0, lane);
#pragma unroll
            for (int n = 0; n < NT; n++)
                acc[m][n] = __builtin_amdgcn_wmma_f32_16x16x32_f16(
                    false, a, false, bf[n], (short)0, acc[m][n], false, false);
        }
    }
    // C/D layout: element r -> row = r + 8*(lane>>4), col = lane&15
#pragma unroll
    for (int n = 0; n < NT; n++) {
        int col = col0 + 16 * n + (lane & 15);
        float bv = bias ? bias[col] : 0.f;
#pragma unroll
        for (int m = 0; m < MT; m++) {
            int rbase = row0 + 16 * m + 8 * (lane >> 4);
#pragma unroll
            for (int r = 0; r < 8; r++) {
                float v = acc[m][n][r] + bv;
                if (RELU) v = v > 0.f ? v : 0.f;
                long idx = (long)(rbase + r) * N + col;
                if (OUTF16) ((_Float16*)Cv)[idx] = (_Float16)v;
                else        ((float*)Cv)[idx] = v;
            }
        }
    }
}

// ---------------- prep kernels ----------------
// Fuse M = (tril(L,-1)+I) @ (triu(U,1)+diag(softplus(d)+1e-3)), store f16.
__global__ void prep_lu(const float* __restrict__ L, const float* __restrict__ U,
                        const float* __restrict__ dvec, _Float16* __restrict__ M) {
    int j = blockIdx.x;       // layer
    int n = blockIdx.y;       // output row
    int k = threadIdx.x;      // output col
    const float* Lj = L + (long)j * DDIM * DDIM;
    const float* Uj = U + (long)j * DDIM * DDIM;
    const float* dj = dvec + j * DDIM;
    float s = 0.f;
    int im = n < k ? n : k;
    for (int i = 0; i <= im; i++) {
        float lo = (i < n) ? Lj[n * DDIM + i] : 1.0f;
        float up = (i < k) ? Uj[i * DDIM + k] : (softplusf(dj[i]) + 0.001f);
        s += lo * up;
    }
    M[((long)j * DDIM + n) * DDIM + k] = (_Float16)s;
}

// single scalar: all row-constant log-det contributions
__global__ void prep_ldconst(const float* __restrict__ an_logs,
                             const float* __restrict__ lu_d,
                             float* __restrict__ ldc) {
    float s = 0.f;
    for (int i = 0; i < 2 * DDIM; i++) s += an_logs[i];
    for (int j = 0; j < NLU; j++)
        for (int d = 0; d < DDIM; d++)
            s += logf(softplusf(lu_d[j * DDIM + d]) + 0.001f);
    ldc[0] = s;
}

// zero-pad W1 from K=72 to K=96 and convert to f16
__global__ void pad_w1(const float* __restrict__ W, _Float16* __restrict__ Wp, int layers) {
    long i = (long)blockIdx.x * blockDim.x + threadIdx.x;
    long total = (long)layers * HID * KP1;
    if (i >= total) return;
    int k = (int)(i % KP1);
    long nl = i / KP1;
    int n = (int)(nl % HID);
    long l = nl / HID;
    Wp[i] = (k < AIN) ? (_Float16)W[(l * HID + n) * AIN + k] : (_Float16)0.f;
}

// generic f32 -> f16 copy
__global__ void cvt_f16(const float* __restrict__ src, _Float16* __restrict__ dst, long n) {
    long i = (long)blockIdx.x * blockDim.x + threadIdx.x;
    if (i < n) dst[i] = (_Float16)src[i];
}

__global__ void gather_ctx(const int* __restrict__ labels, const float* __restrict__ emb,
                           float* __restrict__ CTX) {
    int i = blockIdx.x * blockDim.x + threadIdx.x;
    if (i < BATCH * CTXW) {
        int b = i >> 3, c = i & 7;
        CTX[i] = emb[labels[b] * CTXW + c];
    }
}

__global__ void zero_ld(float* __restrict__ LD) {
    int i = blockIdx.x * blockDim.x + threadIdx.x;
    if (i < BATCH) LD[i] = 0.f;
}

// ---------------- elementwise layers ----------------
__global__ void an_apply(float* __restrict__ Y, const float* __restrict__ logs,
                         const float* __restrict__ shift) {
    int i = blockIdx.x * blockDim.x + threadIdx.x;
    if (i < BATCH * DDIM) {
        int d = i & (DDIM - 1);
        Y[i] = Y[i] * expf(logs[d]) + shift[d];
    }
}

// pack [ y[:,idf] | ctx | 0pad ] -> XIN f16 (chunk-local, K=96)
__global__ void pack_xin(const float* __restrict__ Y, const float* __restrict__ CTX,
                         _Float16* __restrict__ XIN, int b0, int idf_off) {
    int r = blockIdx.x;
    int c = threadIdx.x;        // 0..95
    int b = b0 + r;
    float v;
    if (c < HALF)          v = Y[b * DDIM + 2 * c + idf_off];
    else if (c < AIN)      v = CTX[b * CTXW + (c - HALF)];
    else                   v = 0.f;
    XIN[r * KP1 + c] = (_Float16)v;
}

// affine coupling epilogue: one wave per row, in-place on transformed half
__global__ void af_apply(float* __restrict__ Y, const float* __restrict__ PR,
                         float* __restrict__ LD, int b0, int trf_off) {
    int lane = threadIdx.x & 31;
    int wave = threadIdx.x >> 5;
    int r = blockIdx.x * (blockDim.x >> 5) + wave;
    int b = b0 + r;
    float lsum = 0.f;
    for (int t = lane; t < HALF; t += 32) {
        float shift = PR[r * AOUT + t];
        float scale = 1.f / (1.f + expf(-(PR[r * AOUT + HALF + t] + 2.f))) + 0.001f;
        int col = 2 * t + trf_off;
        float xt = Y[b * DDIM + col];
        Y[b * DDIM + col] = xt * scale + shift;
        lsum += logf(scale);
    }
    for (int o = 16; o > 0; o >>= 1) lsum += __shfl_down(lsum, o, 32);
    if (lane == 0) LD[b] += lsum;
}

// rational-quadratic spline for one scalar given its 23 params
__device__ void rq_point(const float* __restrict__ pp, float x,
                         float& yout, float& lout) {
    float w[8], h[8], cw[9], chh[9], d[9];
    float mw = -1e30f, mh = -1e30f;
#pragma unroll
    for (int i = 0; i < 8; i++) {
        w[i] = pp[i]; h[i] = pp[8 + i];
        mw = fmaxf(mw, w[i]); mh = fmaxf(mh, h[i]);
    }
    float sw = 0.f, sh = 0.f;
#pragma unroll
    for (int i = 0; i < 8; i++) {
        w[i] = expf(w[i] - mw); sw += w[i];
        h[i] = expf(h[i] - mh); sh += h[i];
    }
    const float cmb = 1.f - MBW * 8.f;
    float aw = 0.f, ah = 0.f;
    cw[0] = -TBND; chh[0] = -TBND;
#pragma unroll
    for (int i = 0; i < 8; i++) {
        aw += MBW + cmb * w[i] / sw;
        ah += MBW + cmb * h[i] / sh;
        cw[i + 1]  = 2.f * TBND * aw - TBND;
        chh[i + 1] = 2.f * TBND * ah - TBND;
    }
    cw[8] = TBND; chh[8] = TBND;
    d[0] = 1.0f; d[8] = 1.0f;                    // MD + softplus(DCONST) == 1
#pragma unroll
    for (int i = 0; i < 7; i++) d[i + 1] = MDC + softplusf(pp[16 + i]);

    bool inside = (x >= -TBND) && (x <= TBND);
    float xc = fminf(fmaxf(x, -TBND), TBND);
    int idx = 0;
#pragma unroll
    for (int i = 0; i < 9; i++) idx += (xc >= cw[i]) ? 1 : 0;
    idx -= 1; idx = idx < 0 ? 0 : (idx > 7 ? 7 : idx);

    float icw = cw[idx],  iwb = cw[idx + 1] - cw[idx];
    float ich = chh[idx], ihb = chh[idx + 1] - chh[idx];
    float idl = ihb / iwb;
    float d0 = d[idx], d1 = d[idx + 1];
    float th = (xc - icw) / iwb;
    float t1 = th * (1.f - th);
    float den = idl + (d0 + d1 - 2.f * idl) * t1;
    float y = ich + ihb * (idl * th * th + d0 * t1) / den;
    float dn = idl * idl * (d1 * th * th + 2.f * idl * t1 + d0 * (1.f - th) * (1.f - th));
    float ldv = logf(dn) - 2.f * logf(den);
    yout = inside ? y : x;
    lout = inside ? ldv : 0.f;
}

// spline coupling epilogue: one wave per row
__global__ void rq_apply(float* __restrict__ Y, const float* __restrict__ PR,
                         float* __restrict__ LD, int b0, int trf_off) {
    int lane = threadIdx.x & 31;
    int wave = threadIdx.x >> 5;
    int r = blockIdx.x * (blockDim.x >> 5) + wave;
    int b = b0 + r;
    float lsum = 0.f;
    for (int t = lane; t < HALF; t += 32) {
        const float* pp = PR + (long)r * ROUT + t * 23;
        int col = 2 * t + trf_off;
        float x = Y[b * DDIM + col];
        float yo, lo;
        rq_point(pp, x, yo, lo);
        Y[b * DDIM + col] = yo;
        lsum += lo;
    }
    for (int o = 16; o > 0; o >>= 1) lsum += __shfl_down(lsum, o, 32);
    if (lane == 0) LD[b] += lsum;
}

__global__ void finalize(const float* __restrict__ Y, const float* __restrict__ LD,
                         const float* __restrict__ ldc, float* __restrict__ out) {
    int i = blockIdx.x * blockDim.x + threadIdx.x;
    const int ny = BATCH * DDIM;
    if (i < ny) out[i] = Y[i];
    else if (i < ny + BATCH) out[i] = LD[i - ny] + ldc[0];
}

// ---------------- host side ----------------
struct Step { int kind; int j; int p; };   // 0=an 1=lu 2=af 3=rq

extern "C" void kernel_launch(void* const* d_in, const int* in_sizes, int n_in,
                              void* d_out, int out_size, void* d_ws, size_t ws_size,
                              hipStream_t stream) {
    const float* x        = (const float*)d_in[0];
    const int*   labels   = (const int*)  d_in[1];
    const float* emb      = (const float*)d_in[2];
    const float* an_logs  = (const float*)d_in[3];
    const float* an_shift = (const float*)d_in[4];
    const float* lu_L     = (const float*)d_in[5];
    const float* lu_U     = (const float*)d_in[6];
    const float* lu_d     = (const float*)d_in[7];
    const float* lu_b     = (const float*)d_in[8];
    const float* aW1      = (const float*)d_in[9];
    const float* ab1      = (const float*)d_in[10];
    const float* aW2      = (const float*)d_in[11];
    const float* ab2      = (const float*)d_in[12];
    const float* aW3      = (const float*)d_in[13];
    const float* ab3      = (const float*)d_in[14];
    const float* rW1      = (const float*)d_in[15];
    const float* rb1      = (const float*)d_in[16];
    const float* rW2      = (const float*)d_in[17];
    const float* rb2      = (const float*)d_in[18];
    const float* rW3      = (const float*)d_in[19];
    const float* rb3      = (const float*)d_in[20];

    char* base = (char*)d_ws;
    size_t off = 0;
    auto allocB = [&](size_t bytes) {
        char* p = base + off;
        off = (off + bytes + 255) & ~(size_t)255;
        return p;
    };
    float* Yb[2];
    Yb[0]            = (float*)allocB((size_t)BATCH * DDIM * 4);
    Yb[1]            = (float*)allocB((size_t)BATCH * DDIM * 4);
    float* LD        = (float*)allocB((size_t)BATCH * 4);
    float* CTX       = (float*)allocB((size_t)BATCH * CTXW * 4);
    float* LDC       = (float*)allocB(256);
    _Float16* LUMh   = (_Float16*)allocB((size_t)NLU * DDIM * DDIM * 2);
    _Float16* W1Ah   = (_Float16*)allocB((size_t)NAF * HID * KP1 * 2);
    _Float16* W1Rh   = (_Float16*)allocB((size_t)NRQ * HID * KP1 * 2);
    _Float16* W2Ah   = (_Float16*)allocB((size_t)NAF * HID * HID * 2);
    _Float16* W2Rh   = (_Float16*)allocB((size_t)NRQ * HID * HID * 2);
    _Float16* W3Ah   = (_Float16*)allocB((size_t)NAF * AOUT * HID * 2);
    _Float16* W3Rh   = (_Float16*)allocB((size_t)NRQ * ROUT * HID * 2);
    _Float16* XIN    = (_Float16*)allocB((size_t)CH * KP1 * 2);
    _Float16* H1     = (_Float16*)allocB((size_t)CH * HID * 2);
    _Float16* H2     = (_Float16*)allocB((size_t)CH * HID * 2);
    float* PR        = (float*)allocB((size_t)CH * ROUT * 4);
    (void)ws_size; (void)in_sizes; (void)n_in; (void)out_size;

    // ---- prep ----
    hipMemcpyAsync(Yb[0], x, (size_t)BATCH * DDIM * sizeof(float),
                   hipMemcpyDeviceToDevice, stream);
    zero_ld<<<(BATCH + 255) / 256, 256, 0, stream>>>(LD);
    gather_ctx<<<(BATCH * CTXW + 255) / 256, 256, 0, stream>>>(labels, emb, CTX);
    prep_lu<<<dim3(NLU, DDIM), DDIM, 0, stream>>>(lu_L, lu_U, lu_d, LUMh);
    prep_ldconst<<<1, 1, 0, stream>>>(an_logs, lu_d, LDC);
    {
        long ta = (long)NAF * HID * KP1, tr = (long)NRQ * HID * KP1;
        pad_w1<<<(int)((ta + 255) / 256), 256, 0, stream>>>(aW1, W1Ah, NAF);
        pad_w1<<<(int)((tr + 255) / 256), 256, 0, stream>>>(rW1, W1Rh, NRQ);
        auto cvt = [&](const float* s, _Float16* d, long n) {
            cvt_f16<<<(int)((n + 255) / 256), 256, 0, stream>>>(s, d, n);
        };
        cvt(aW2, W2Ah, (long)NAF * HID * HID);
        cvt(rW2, W2Rh, (long)NRQ * HID * HID);
        cvt(aW3, W3Ah, (long)NAF * AOUT * HID);
        cvt(rW3, W3Rh, (long)NRQ * ROUT * HID);
    }

    // ---- schedule (mirrors reference _schedule) ----
    std::vector<Step> sch;
    { int an = 0, lu = 0, af = 0, rq = 0;
      for (int i = 0; i < 2; i++) {
          sch.push_back({0, an++, 0});
          sch.push_back({1, lu++, 0});
          sch.push_back({2, af++, i % 2});
      }
      for (int i = 0; i < 4; i++) {
          sch.push_back({1, lu++, 0});
          sch.push_back({2, af++, i % 2});
          for (int k = 0; k < 3; k++) {
              sch.push_back({1, lu++, 0});
              sch.push_back({3, rq++, i % 2});
          }
      }
      for (int i = 0; i < 3; i++) {
          sch.push_back({1, lu++, 0});
          sch.push_back({3, rq++, i % 2});
      }
      sch.push_back({1, lu++, 0});
    }

    int cur = 0;
    const int NCHUNK = BATCH / CH;
    for (const Step& s : sch) {
        if (s.kind == 0) {                       // an
            an_apply<<<(BATCH * DDIM + 255) / 256, 256, 0, stream>>>(
                Yb[cur], an_logs + s.j * DDIM, an_shift + s.j * DDIM);
        } else if (s.kind == 1) {                // lu : y' = y @ M^T + b
            // wave tile 32x64; block = 4 waves = 128 rows; grid.y = 128/64 = 2
            gemm_wmma<0, 1, 0, 2, 4><<<dim3(BATCH / 128, DDIM / 64), 128, 0, stream>>>(
                Yb[cur], LUMh + (size_t)s.j * DDIM * DDIM, lu_b + s.j * DDIM,
                Yb[1 - cur], DDIM, DDIM);
            cur ^= 1;
        } else {                                 // af / rq coupling
            bool is_af = (s.kind == 2);
            int idf_off = (s.p == 0) ? 0 : 1;    // identity half parity
            int trf_off = (s.p == 0) ? 1 : 0;    // transformed half parity
            const _Float16* W1 = is_af ? (W1Ah + (size_t)s.j * HID * KP1)
                                       : (W1Rh + (size_t)s.j * HID * KP1);
            const float* b1 = (is_af ? ab1 : rb1) + s.j * HID;
            const _Float16* W2 = (is_af ? W2Ah : W2Rh) + (size_t)s.j * HID * HID;
            const float* b2 = (is_af ? ab2 : rb2) + s.j * HID;
            int N3 = is_af ? AOUT : ROUT;
            const _Float16* W3 = is_af ? (W3Ah + (size_t)s.j * AOUT * HID)
                                       : (W3Rh + (size_t)s.j * ROUT * HID);
            const float* b3 = is_af ? (ab3 + s.j * AOUT) : (rb3 + s.j * ROUT);
            for (int c = 0; c < NCHUNK; c++) {
                int b0 = c * CH;
                pack_xin<<<CH, KP1, 0, stream>>>(Yb[cur], CTX, XIN, b0, idf_off);
                gemm_wmma<1, 0, 1, 2, 4><<<dim3(CH / 128, HID / 64), 128, 0, stream>>>(
                    XIN, W1, b1, H1, KP1, HID);
                gemm_wmma<1, 0, 1, 2, 4><<<dim3(CH / 128, HID / 64), 128, 0, stream>>>(
                    H1, W2, b2, H2, HID, HID);
                gemm_wmma<0, 0, 0, 2, 4><<<dim3(CH / 128, N3 / 64), 128, 0, stream>>>(
                    H2, W3, b3, PR, HID, N3);
                if (is_af)
                    af_apply<<<CH / 8, 256, 0, stream>>>(Yb[cur], PR, LD, b0, trf_off);
                else
                    rq_apply<<<CH / 8, 256, 0, stream>>>(Yb[cur], PR, LD, b0, trf_off);
            }
        }
    }

    finalize<<<(BATCH * DDIM + BATCH + 255) / 256, 256, 0, stream>>>(
        Yb[cur], LD, LDC, (float*)d_out);
}
